// RPN_54924041781267
// MI455X (gfx1250) — compile-verified, matrix operand order checked by reading, not run on hardware
//
#include <hip/hip_runtime.h>
#include <hip/hip_bf16.h>
#include <math.h>

typedef __attribute__((ext_vector_type(16))) _Float16 v16h;
typedef __attribute__((ext_vector_type(8)))  float    v8f;

#define A_IMG    36864
#define A_PAD    65536
#define PRE_NMS  2000
#define POST_NMS 300
#define NMS_TH   0.7f
#define MIN_SZ   16.0f

// ---- workspace layout (bytes) ----
#define OFF_FEATP   0ull
#define SZ_FEATP    (8ull*66*66*512*2)        // padded NHWC f16 feat
#define OFF_WCONV   (OFF_FEATP + SZ_FEATP)
#define SZ_WCONV    (512ull*4608*2)           // conv weights [o][k] f16
#define OFF_CONVOUT (OFF_WCONV + SZ_WCONV)
#define SZ_CONVOUT  (32768ull*512*2)          // conv output f16
#define OFF_WHEAD   (OFF_CONVOUT + SZ_CONVOUT)
#define SZ_WHEAD    (64ull*512*2)             // head weights [o64][c] f16
#define OFF_BOXES   (OFF_WHEAD + SZ_WHEAD)
#define SZ_BOXES    (8ull*36864*4*4)          // decoded boxes f32
#define OFF_KEYS    (OFF_BOXES + SZ_BOXES)
#define SZ_KEYS     (8ull*65536*8)            // sort keys u64
#define OFF_TBOX    (OFF_KEYS + SZ_KEYS)
#define SZ_TBOX     (8ull*2000*4*4)           // top-2000 boxes
#define OFF_KEEP    (OFF_TBOX + SZ_TBOX)
#define SZ_KEEP     (8ull*2000*4)             // keep flags

// ---- CDNA5 async global->LDS path (probe-guarded) ----
#if defined(__HIP_DEVICE_COMPILE__) && __has_builtin(__builtin_amdgcn_global_load_async_to_lds_b128) && __has_builtin(__builtin_amdgcn_s_wait_asynccnt)
#define USE_ASYNC_LDS 1
#else
#define USE_ASYNC_LDS 0
#endif

#if USE_ASYNC_LDS
typedef __attribute__((ext_vector_type(4))) int vi4;
typedef __attribute__((address_space(1))) vi4 gvi4;   // global int4
typedef __attribute__((address_space(3))) vi4 lvi4;   // LDS int4
#endif

__device__ inline void stage16(_Float16* dst, const _Float16* src) {
#if USE_ASYNC_LDS
  __builtin_amdgcn_global_load_async_to_lds_b128(
      (gvi4*)(void*)const_cast<_Float16*>(src),
      (lvi4*)(void*)dst, 0, 0);
#else
  *(uint4*)dst = *(const uint4*)src;
#endif
}

__device__ inline void stage_wait() {
#if USE_ASYNC_LDS
  __builtin_amdgcn_s_wait_asynccnt(0);
#endif
}

__device__ inline v8f wmma16(v16h a, v16h b, v8f c) {
  return __builtin_amdgcn_wmma_f32_16x16x32_f16(false, a, false, b, (short)0, c, false, false);
}

union Frag { v16h v; uint4 q[2]; };

// ============ layout-prep kernels ============
__global__ void k_zero_half(_Float16* p, unsigned n16) {
  unsigned i = blockIdx.x*blockDim.x + threadIdx.x;
  if (i < n16) ((uint4*)p)[i] = make_uint4(0u,0u,0u,0u);
}

__global__ void k_feat_nhwc(const float* __restrict__ feat, _Float16* __restrict__ featP) {
  int i = blockIdx.x*blockDim.x + threadIdx.x;         // 8*512*64*64
  int x = i & 63, y = (i>>6)&63, c = (i>>12)&511, b = i>>21;
  featP[((size_t)((b*66 + y+1)*66 + (x+1)))*512 + c] = (_Float16)feat[i];
}

__global__ void k_wconv(const float* __restrict__ w, _Float16* __restrict__ wT) {
  int i = blockIdx.x*blockDim.x + threadIdx.x;         // 512*512*9
  int kx = i % 3; int t = i / 3; int ky = t % 3; t /= 3;
  int c = t & 511; int o = t >> 9;
  wT[(size_t)o*4608 + (ky*3+kx)*512 + c] = (_Float16)w[i];
}

__global__ void k_whead(const float* __restrict__ locw, const float* __restrict__ clsw,
                        _Float16* __restrict__ wh) {
  int i = blockIdx.x*blockDim.x + threadIdx.x;         // 64*512
  int o = i >> 9, c = i & 511;
  float v = 0.f;
  if (o < 36)      v = locw[o*512 + c];
  else if (o < 54) v = clsw[(o-36)*512 + c];
  wh[i] = (_Float16)v;
}

__global__ void k_anchors(float* __restrict__ anch) {
  int i = blockIdx.x*blockDim.x + threadIdx.x;         // 36864
  if (i >= A_IMG) return;
  int k = i % 9, pix = i / 9;
  int x = pix & 63, y = pix >> 6;
  int r = k / 3, s = k % 3;
  const float RAT[3] = {0.5f, 1.f, 2.f};
  const float SC[3]  = {8.f, 16.f, 32.f};
  float hh = 16.f * SC[s] * sqrtf(RAT[r]);
  float ww = 16.f * SC[s] * sqrtf(1.f/RAT[r]);
  float cx = x*16.f + 8.f, cy = y*16.f + 8.f;
  anch[i*4+0] = cx - 0.5f*ww;
  anch[i*4+1] = cy - 0.5f*hh;
  anch[i*4+2] = cx + 0.5f*ww;
  anch[i*4+3] = cy + 0.5f*hh;
}

// ============ 3x3 conv as implicit GEMM (WMMA f16, double-buffered async LDS) ====
// M=32768 (b,y,x) x N=512 (o), K=4608 (tap*512+c). Tile 64x128 per block.
struct StageCtx {
  const _Float16* featP;
  const _Float16* wT;
  int ab, ay, ax, arow, apart, tid, tileN;
};

__device__ inline void stage_tiles(const StageCtx& c, int s,
                                   _Float16 (*As)[40], _Float16 (*Bs)[40]) {
  const int tap = s >> 4;              // 0..8 (ky*3+kx)
  const int c0  = (s & 15) << 5;       // channel base within tap
  const int dy = tap / 3, dx = tap % 3;
  // A tile (64 rows x 32 K)
  const _Float16* srcA = c.featP +
    ((size_t)((c.ab*66 + c.ay + dy)*66 + (c.ax + dx)))*512 + c0 + c.apart*8;
  stage16(&As[c.arow][c.apart*8], srcA);
  // B tile (128 cols x 32 K)
  for (int r = 0; r < 2; ++r) {
    int j = c.tid + r*256;
    int col = j >> 2, part = j & 3;
    const _Float16* srcB = c.wT +
      (size_t)(c.tileN*128 + col)*4608 + tap*512 + c0 + part*8;
    stage16(&Bs[col][part*8], srcB);
  }
}

__global__ __launch_bounds__(256) void k_conv_gemm(
    const _Float16* __restrict__ featP, const _Float16* __restrict__ wT,
    const float* __restrict__ bias, _Float16* __restrict__ out)
{
  __shared__ __align__(16) _Float16 As[2][64][40];   // double-buffered, pad 40
  __shared__ __align__(16) _Float16 Bs[2][128][40];

  const int tid  = threadIdx.x;
  const int lane = tid & 31;
  const int wave = tid >> 5;
  const int wm = wave >> 2;            // 0..1 (M sub-tile)
  const int wn = wave & 3;             // 0..3 (N sub-tile)
  const int tileM = blockIdx.x;        // 0..511
  const int tileN = blockIdx.y;        // 0..3

  const int arow  = tid >> 2;
  const int apart = tid & 3;
  const int m_g = tileM*64 + arow;

  StageCtx ctx;
  ctx.featP = featP; ctx.wT = wT;
  ctx.ab = m_g >> 12; ctx.ay = (m_g >> 6) & 63; ctx.ax = m_g & 63;
  ctx.arow = arow; ctx.apart = apart; ctx.tid = tid; ctx.tileN = tileN;

  v8f acc00 = {}; v8f acc01 = {}; v8f acc10 = {}; v8f acc11 = {};

  const int lrow = lane & 15;
  const int kb   = (lane >> 4) << 3;   // 0 or 8 (K sub-block per ISA layout)

  // prologue: stage step 0 into buffer 0
  stage_tiles(ctx, 0, As[0], Bs[0]);

  for (int s = 0; s < 144; ++s) {
    const int cur = s & 1;
    stage_wait();          // own async loads for 'cur' landed
    __syncthreads();       // all waves' loads landed; all reads of 'cur^1' done
    if (s + 1 < 144) stage_tiles(ctx, s + 1, As[cur ^ 1], Bs[cur ^ 1]);

    Frag a0, a1, b0, b1;
    const _Float16* ap0 = &As[cur][wm*32 + lrow][kb];
    a0.q[0] = *(const uint4*)ap0;       a0.q[1] = *(const uint4*)(ap0 + 16);
    const _Float16* ap1 = &As[cur][wm*32 + 16 + lrow][kb];
    a1.q[0] = *(const uint4*)ap1;       a1.q[1] = *(const uint4*)(ap1 + 16);
    const _Float16* bp0 = &Bs[cur][wn*32 + lrow][kb];
    b0.q[0] = *(const uint4*)bp0;       b0.q[1] = *(const uint4*)(bp0 + 16);
    const _Float16* bp1 = &Bs[cur][wn*32 + 16 + lrow][kb];
    b1.q[0] = *(const uint4*)bp1;       b1.q[1] = *(const uint4*)(bp1 + 16);

    acc00 = wmma16(a0.v, b0.v, acc00);
    acc01 = wmma16(a0.v, b1.v, acc01);
    acc10 = wmma16(a1.v, b0.v, acc10);
    acc11 = wmma16(a1.v, b1.v, acc11);
  }

  // epilogue: bias + ReLU, store f16
  const int mrow0 = tileM*64 + wm*32;
  const int ncol0 = tileN*128 + wn*32;
  const int rofs  = (lane >> 4) << 3;
  const int ncol  = lane & 15;
  for (int ti = 0; ti < 2; ++ti)
    for (int tj = 0; tj < 2; ++tj) {
      v8f acc = (ti==0) ? (tj==0 ? acc00 : acc01) : (tj==0 ? acc10 : acc11);
      int o = ncol0 + tj*16 + ncol;
      float bv = bias[o];
      for (int r = 0; r < 8; ++r) {
        int m = mrow0 + ti*16 + rofs + r;
        float v = acc[r] + bv;
        v = v > 0.f ? v : 0.f;
        out[(size_t)m*512 + o] = (_Float16)v;
      }
    }
}

// ============ head 1x1 convs (WMMA, N=64 = 36 loc + 18 cls + pad) ============
__global__ __launch_bounds__(256) void k_head_gemm(
    const _Float16* __restrict__ convout, const _Float16* __restrict__ whead,
    const float* __restrict__ loc_b, const float* __restrict__ cls_b,
    float* __restrict__ out_loc, float* __restrict__ out_sc)
{
  const int tid = threadIdx.x, lane = tid & 31, wave = tid >> 5;
  const int mbase = blockIdx.x*128 + wave*16;
  const int lrow = lane & 15;
  const int kb   = (lane >> 4) << 3;

  v8f acc[4] = {{}, {}, {}, {}};
  const _Float16* arow = convout + (size_t)(mbase + lrow)*512;

  for (int kk = 0; kk < 16; ++kk) {
    Frag a;
    const _Float16* ap = arow + kk*32 + kb;
    a.q[0] = *(const uint4*)ap; a.q[1] = *(const uint4*)(ap + 16);
    for (int tj = 0; tj < 4; ++tj) {
      Frag bf;
      const _Float16* bp = whead + (size_t)(tj*16 + lrow)*512 + kk*32 + kb;
      bf.q[0] = *(const uint4*)bp; bf.q[1] = *(const uint4*)(bp + 16);
      acc[tj] = wmma16(a.v, bf.v, acc[tj]);
    }
  }

  const int rofs = (lane >> 4) << 3;
  for (int tj = 0; tj < 4; ++tj) {
    int col = tj*16 + lrow;
    for (int r = 0; r < 8; ++r) {
      int m = mbase + rofs + r;
      float v = acc[tj][r];
      if (col < 36)      out_loc[(size_t)m*36 + col]       = v + loc_b[col];
      else if (col < 54) out_sc [(size_t)m*18 + (col-36)]  = v + cls_b[col-36];
    }
  }
}

// ============ fg + decode + clip + minsize + sort keys ============
__global__ void k_decode(const float* __restrict__ loc, const float* __restrict__ sc,
                         const float* __restrict__ anch,
                         const int* __restrict__ pimg_h, const int* __restrict__ pimg_w,
                         float* __restrict__ boxes, unsigned long long* __restrict__ keys)
{
  int i = blockIdx.x*blockDim.x + threadIdx.x;   // 8 * 65536
  int b = i >> 16, a = i & 65535;
  if (a >= A_IMG) { keys[i] = 0ull; return; }
  int pix = a / 9, k = a % 9;
  int m = b*4096 + pix;

  float s0 = sc[(size_t)m*18 + 2*k], s1 = sc[(size_t)m*18 + 2*k + 1];
  float fg = 1.f / (1.f + expf(s0 - s1));

  float ax0 = anch[a*4+0], ay0 = anch[a*4+1], ax1 = anch[a*4+2], ay1 = anch[a*4+3];
  float wa = ax1 - ax0, ha = ay1 - ay0;
  float cxa = ax0 + 0.5f*wa, cya = ay0 + 0.5f*ha;
  const float* lp = loc + (size_t)m*36 + 4*k;
  float dx = lp[0], dy = lp[1], dw = lp[2], dh = lp[3];
  float cx = dx*wa + cxa, cy = dy*ha + cya;
  float w = expf(dw)*wa, h = expf(dh)*ha;
  float iw = (float)pimg_w[0], ih = (float)pimg_h[0];
  float x0 = fminf(fmaxf(cx - 0.5f*w, 0.f), iw);
  float y0 = fminf(fmaxf(cy - 0.5f*h, 0.f), ih);
  float x1 = fminf(fmaxf(cx + 0.5f*w, 0.f), iw);
  float y1 = fminf(fmaxf(cy + 0.5f*h, 0.f), ih);

  size_t bo = ((size_t)b*A_IMG + a)*4;
  boxes[bo+0]=x0; boxes[bo+1]=y0; boxes[bo+2]=x1; boxes[bo+3]=y1;

  bool ok = (x1 - x0 >= MIN_SZ) && (y1 - y0 >= MIN_SZ);
  float score = ok ? fg : -INFINITY;
  unsigned u = __float_as_uint(score);
  unsigned mapped = (u & 0x80000000u) ? ~u : (u | 0x80000000u);   // monotone map
  keys[i] = ((unsigned long long)mapped << 32) |
            (unsigned long long)(0xFFFFFFFFu - (unsigned)a);      // tie: lower idx first
}

// ============ per-image bitonic sort (descending) over 65536 keys ============
__global__ __launch_bounds__(1024) void k_sort(unsigned long long* __restrict__ keys) {
  unsigned long long* k = keys + ((size_t)blockIdx.x << 16);
  const int N = 65536;
  for (int size = 2; size <= N; size <<= 1) {
    for (int stride = size >> 1; stride > 0; stride >>= 1) {
      __syncthreads();
      for (int t = threadIdx.x; t < (N >> 1); t += 1024) {
        int lo = 2*t - (t & (stride - 1));
        int hi = lo + stride;
        unsigned long long va = k[lo], vb = k[hi];
        bool desc = ((lo & size) == 0);
        bool sw = desc ? (va < vb) : (va > vb);
        if (sw) { k[lo] = vb; k[hi] = va; }
      }
    }
  }
}

__device__ inline float iou4(float4 a, float4 b) {
  float aa = (a.z - a.x)*(a.w - a.y);
  float ab = (b.z - b.x)*(b.w - b.y);
  float lx = fmaxf(a.x, b.x), ly = fmaxf(a.y, b.y);
  float rx = fminf(a.z, b.z), ry = fminf(a.w, b.w);
  float w = fmaxf(rx - lx, 0.f), h = fmaxf(ry - ly, 0.f);
  float inter = w*h;
  return inter / (aa + ab - inter + 1e-9f);
}

// ============ greedy NMS over top-2000, one block per image ============
__global__ __launch_bounds__(256) void k_nms(const unsigned long long* __restrict__ keys,
                                             const float* __restrict__ boxes,
                                             float* __restrict__ tbox, int* __restrict__ keepg)
{
  __shared__ float4 sb[PRE_NMS];
  __shared__ int skeep[PRE_NMS];
  __shared__ int sflag;
  int b = blockIdx.x, tid = threadIdx.x;

  for (int i = tid; i < PRE_NMS; i += 256) {
    unsigned long long kk = keys[((size_t)b << 16) + i];
    unsigned hi  = (unsigned)(kk >> 32);
    unsigned idx = 0xFFFFFFFFu - (unsigned)(kk & 0xFFFFFFFFu);
    float4 bx = make_float4(0.f, 0.f, 0.f, 0.f);
    if (idx < A_IMG) {
      const float* bp = boxes + ((size_t)b*A_IMG + idx)*4;
      bx = make_float4(bp[0], bp[1], bp[2], bp[3]);
    }
    sb[i] = bx;
    skeep[i] = (hi >= 0x80000000u) ? 1 : 0;   // score > -inf
  }
  __syncthreads();

  for (int i = 1; i < PRE_NMS; ++i) {
    if (tid == 0) sflag = 0;
    __syncthreads();
    if (skeep[i]) {
      float4 bi = sb[i];
      int f = 0;
      for (int j = tid; j < i; j += 256)
        if (skeep[j] && iou4(bi, sb[j]) > NMS_TH) { f = 1; break; }
      if (f) sflag = 1;
    }
    __syncthreads();
    if (tid == 0 && sflag) skeep[i] = 0;
    __syncthreads();
  }

  for (int i = tid; i < PRE_NMS; i += 256) {
    float4 bx = sb[i];
    size_t o = ((size_t)b*PRE_NMS + i)*4;
    tbox[o+0]=bx.x; tbox[o+1]=bx.y; tbox[o+2]=bx.z; tbox[o+3]=bx.w;
    keepg[b*PRE_NMS + i] = skeep[i];
  }
}

// ============ compact kept boxes -> 300 rois + indices ============
__global__ __launch_bounds__(256) void k_finalize(const int* __restrict__ keepg,
                                                  const float* __restrict__ tbox,
                                                  float* __restrict__ rois,
                                                  float* __restrict__ ridx)
{
  int b = blockIdx.x, tid = threadIdx.x;
  for (int i = tid; i < POST_NMS*4; i += 256) rois[b*POST_NMS*4 + i] = 0.f;
  for (int i = tid; i < POST_NMS;   i += 256) ridx[b*POST_NMS + i] = (float)b;
  __syncthreads();
  if (tid == 0) {
    int rank = 0;
    for (int i = 0; i < PRE_NMS && rank < POST_NMS; ++i) {
      if (keepg[b*PRE_NMS + i]) {
        const float* s = tbox + ((size_t)b*PRE_NMS + i)*4;
        float* d = rois + ((size_t)b*POST_NMS + rank)*4;
        d[0]=s[0]; d[1]=s[1]; d[2]=s[2]; d[3]=s[3];
        ++rank;
      }
    }
  }
}

extern "C" void kernel_launch(void* const* d_in, const int* in_sizes, int n_in,
                              void* d_out, int out_size, void* d_ws, size_t ws_size,
                              hipStream_t stream)
{
  const float* feat   = (const float*)d_in[0];
  const float* conv1w = (const float*)d_in[1];
  const float* conv1b = (const float*)d_in[2];
  const float* clsw   = (const float*)d_in[3];
  const float* clsb   = (const float*)d_in[4];
  const float* locw   = (const float*)d_in[5];
  const float* locb   = (const float*)d_in[6];
  const int*   imgh   = (const int*)d_in[7];
  const int*   imgw   = (const int*)d_in[8];

  char* ws = (char*)d_ws;
  _Float16* featP   = (_Float16*)(ws + OFF_FEATP);
  _Float16* wT      = (_Float16*)(ws + OFF_WCONV);
  _Float16* convout = (_Float16*)(ws + OFF_CONVOUT);
  _Float16* whead   = (_Float16*)(ws + OFF_WHEAD);
  float*    boxes   = (float*)(ws + OFF_BOXES);
  unsigned long long* keys = (unsigned long long*)(ws + OFF_KEYS);
  float*    tbox    = (float*)(ws + OFF_TBOX);
  int*      keepg   = (int*)(ws + OFF_KEEP);

  float* out    = (float*)d_out;
  float* o_loc  = out;                 // (8,36864,4)  = 1,179,648
  float* o_sc   = out + 1179648;       // (8,36864,2)  =   589,824
  float* o_roi  = out + 1769472;       // (2400,4)     =     9,600
  float* o_ridx = out + 1779072;       // (2400,)      =     2,400
  float* o_anch = out + 1781472;       // (36864,4)    =   147,456

  k_zero_half<<<(unsigned)(SZ_FEATP/16/256), 256, 0, stream>>>(featP, (unsigned)(SZ_FEATP/16));
  k_feat_nhwc<<<65536, 256, 0, stream>>>(feat, featP);
  k_wconv<<<9216, 256, 0, stream>>>(conv1w, wT);
  k_whead<<<128, 256, 0, stream>>>(locw, clsw, whead);
  k_anchors<<<144, 256, 0, stream>>>(o_anch);

  dim3 cg(512, 4);
  k_conv_gemm<<<cg, 256, 0, stream>>>(featP, wT, conv1b, convout);
  k_head_gemm<<<256, 256, 0, stream>>>(convout, whead, locb, clsb, o_loc, o_sc);

  k_decode<<<2048, 256, 0, stream>>>(o_loc, o_sc, o_anch, imgh, imgw, boxes, keys);
  k_sort<<<8, 1024, 0, stream>>>(keys);
  k_nms<<<8, 256, 0, stream>>>(keys, boxes, tbox, keepg);
  k_finalize<<<8, 256, 0, stream>>>(keepg, tbox, o_roi, o_ridx);
}
